// VCSA_23536420782399
// MI455X (gfx1250) — compile-verified, matrix operand order checked by reading, not run on hardware
//
#include <hip/hip_runtime.h>
#include <hip/hip_bf16.h>

typedef __attribute__((ext_vector_type(16))) _Float16 v16h;
typedef __attribute__((ext_vector_type(8)))  _Float16 v8h;
typedef __attribute__((ext_vector_type(8)))  float    v8f;
typedef __attribute__((ext_vector_type(4))) unsigned int u32x4;
typedef __attribute__((ext_vector_type(8)))  int i32x8;
typedef __attribute__((ext_vector_type(4)))  int i32x4;

#define N_EDGES_C  800000
#define N_NODES_C  50000
#define NODE_DIM_C 128
#define HIDDEN_C   64
// 32 fragments (4 ntiles x 8 kchunks), each: 2 chunks x 32 lanes x 8 halfs
#define W1_FRAG_ELEMS (32 * 2 * 32 * 8)          // 16384 f16 = 32KB
#define W1_FRAG_BYTES ((size_t)W1_FRAG_ELEMS * 2)
#define XH_BYTES      ((size_t)N_NODES_C * NODE_DIM_C * 2)

// ---------------------------------------------------------------------------
// Pre-pass 1: pack W1 (256 x 64 f32, row-major) into f16 WMMA B-fragment
// order, chunk-major so LDS reads are two coalesced ds_load_b128:
//   elem offset = frag*512 + chunk*256 + lane*8 + jj     (frag = nt*8 + kc)
//   B element  K = kc*32 + chunk*16 + jj + (lane>=16 ? 8 : 0),
//              N = nt*16 + lane%16
// ---------------------------------------------------------------------------
__global__ void VCSA_pack_w1(const float* __restrict__ W1,
                             _Float16* __restrict__ wsB) {
    int idx = blockIdx.x * blockDim.x + threadIdx.x;
    if (idx >= W1_FRAG_ELEMS) return;
    int jj    = idx & 7;
    int l     = (idx >> 3) & 31;
    int chunk = (idx >> 8) & 1;
    int frag  = idx >> 9;
    int kc    = frag & 7;
    int nt    = frag >> 3;
    int k = kc * 32 + chunk * 16 + jj + ((l >= 16) ? 8 : 0);
    int n = nt * 16 + (l & 15);
    wsB[idx] = (_Float16)W1[k * HIDDEN_C + n];
}

// ---------------------------------------------------------------------------
// Pre-pass 2: convert node features f32 -> f16 once (each row is gathered
// ~32x by edges; convert-once removes all hot-loop cvts & halves gather bytes)
// ---------------------------------------------------------------------------
__global__ void VCSA_pack_x(const float* __restrict__ x,
                            _Float16* __restrict__ xh, int n8) {
    int i = blockIdx.x * blockDim.x + threadIdx.x;   // one thread per 8 elems
    if (i >= n8) return;
    const float4 f0 = *(const float4*)(x + (size_t)i * 8);
    const float4 f1 = *(const float4*)(x + (size_t)i * 8 + 4);
    v8h h;
    h[0] = (_Float16)f0.x; h[1] = (_Float16)f0.y;
    h[2] = (_Float16)f0.z; h[3] = (_Float16)f0.w;
    h[4] = (_Float16)f1.x; h[5] = (_Float16)f1.y;
    h[6] = (_Float16)f1.z; h[7] = (_Float16)f1.w;
    *(v8h*)(xh + (size_t)i * 8) = h;
}

// ---------------------------------------------------------------------------
// Shared device helpers
// ---------------------------------------------------------------------------
__device__ __forceinline__ void VCSA_tdm_fill_lds(const _Float16* wsB, int wave,
                                                  _Float16* ldsB) {
    if (wave == 0) {
        const unsigned long long ga = (unsigned long long)wsB;
        u32x4 g0;
        g0[0] = 1u;                                   // count=1, user mode
        g0[1] = 0u;                                   // lds_addr = 0 (ldsB base)
        g0[2] = (unsigned int)(ga & 0xFFFFFFFFull);   // global_addr[31:0]
        g0[3] = (unsigned int)((ga >> 32) & 0x01FFFFFFull) | (2u << 30); // type=2
        i32x8 g1;
        g1[0] = (int)(2u << 16);       // data_size=4B
        g1[1] = (int)(8192u << 16);    // tensor_dim0[15:0]=8192
        g1[2] = (int)(1u << 16);       // tensor_dim1=1
        g1[3] = (int)(8192u << 16);    // tile_dim0=8192
        g1[4] = 1;                     // tile_dim1=1
        g1[5] = 8192;                  // tensor_dim0_stride
        g1[6] = 0;
        g1[7] = 0;
        i32x4 gz;  gz[0] = gz[1] = gz[2] = gz[3] = 0;
        i32x8 gz8; gz8[0] = gz8[1] = gz8[2] = gz8[3] = 0;
                   gz8[4] = gz8[5] = gz8[6] = gz8[7] = 0;
        __builtin_amdgcn_tensor_load_to_lds(g0, g1, gz, gz, gz8, 0);
        __builtin_amdgcn_s_wait_tensorcnt(0);
    }
    // TDM wrote LDS behind the compiler's back; escape the address so the
    // "never-stored LDS global" folding cannot eliminate the ds loads.
    _Float16* p = ldsB;
    asm volatile("" : "+v"(p) : : "memory");
    __syncthreads();
}

__device__ __forceinline__ void VCSA_layer2(const v8f c[4], const float* b1,
                                            const float* W2, const float* b2,
                                            int m, int hi, int ebase,
                                            float* out) {
    float b1v[4], w2v[4];
#pragma unroll
    for (int nt = 0; nt < 4; ++nt) {
        b1v[nt] = b1[nt * 16 + m];
        w2v[nt] = W2[nt * 16 + m];
    }
    const float bias2 = b2[0];
#pragma unroll
    for (int r = 0; r < 8; ++r) {
        float partial = 0.0f;
#pragma unroll
        for (int nt = 0; nt < 4; ++nt) {
            float h = c[nt][r] + b1v[nt];
            h = (h > 0.0f) ? h : 0.0f;
            partial += h * w2v[nt];
        }
        partial += __shfl_xor(partial, 1, 32);
        partial += __shfl_xor(partial, 2, 32);
        partial += __shfl_xor(partial, 4, 32);
        partial += __shfl_xor(partial, 8, 32);
        if (m == r) {
            const float z = partial + bias2;
            out[ebase + r + hi] = 1.0f / (1.0f + __expf(-z));
        }
    }
}

// ---------------------------------------------------------------------------
// Main (f16 gather path): one wave32 per 16-edge tile, 32 WMMAs/tile.
// ---------------------------------------------------------------------------
__global__ __launch_bounds__(256) void VCSA_edge_mlp_h(
    const _Float16* __restrict__ xh,   // f16 node features [N,128]
    const int*   __restrict__ ei,
    const _Float16* __restrict__ wsB,
    const float* __restrict__ b1,
    const float* __restrict__ W2,
    const float* __restrict__ b2,
    float*       __restrict__ out)
{
    __shared__ _Float16 ldsB[W1_FRAG_ELEMS];   // 32KB at LDS offset 0
    const int lane = threadIdx.x & 31;
    const int wave = threadIdx.x >> 5;
    VCSA_tdm_fill_lds(wsB, wave, ldsB);

    const int tile  = blockIdx.x * 8 + wave;
    const int ebase = tile * 16;
    const int m     = lane & 15;
    const int hi    = (lane >= 16) ? 8 : 0;

    const int e    = ebase + m;
    const int nrow = ei[e];
    const int ncol = ei[N_EDGES_C + e];
    const _Float16* __restrict__ xr = xh + (size_t)nrow * NODE_DIM_C;
    const _Float16* __restrict__ xc = xh + (size_t)ncol * NODE_DIM_C;

    v8f c[4];
#pragma unroll
    for (int nt = 0; nt < 4; ++nt)
#pragma unroll
        for (int r = 0; r < 8; ++r) c[nt][r] = 0.0f;

#pragma unroll
    for (int kc = 0; kc < 8; ++kc) {
        const int k1 = kc * 32 + hi;
        const int k2 = k1 + 16;
        const _Float16* p1 = (k1 < NODE_DIM_C) ? (xr + k1) : (xc + (k1 - NODE_DIM_C));
        const _Float16* p2 = (k2 < NODE_DIM_C) ? (xr + k2) : (xc + (k2 - NODE_DIM_C));
        const v8h r1 = *(const v8h*)p1;        // one global_load_b128 each
        const v8h r2 = *(const v8h*)p2;
        const v16h a = __builtin_shufflevector(r1, r2,
            0, 1, 2, 3, 4, 5, 6, 7, 8, 9, 10, 11, 12, 13, 14, 15);

        v16h bfr[4];
#pragma unroll
        for (int nt = 0; nt < 4; ++nt) {
            const int fo = (nt * 8 + kc) * 512;
            const v8h blo = *(const v8h*)(ldsB + fo + lane * 8);
            const v8h bhi = *(const v8h*)(ldsB + fo + 256 + lane * 8);
            bfr[nt] = __builtin_shufflevector(blo, bhi,
                0, 1, 2, 3, 4, 5, 6, 7, 8, 9, 10, 11, 12, 13, 14, 15);
        }
#pragma unroll
        for (int nt = 0; nt < 4; ++nt) {
            c[nt] = __builtin_amdgcn_wmma_f32_16x16x32_f16(
                false, a, false, bfr[nt], (short)0, c[nt], false, false);
        }
    }
    VCSA_layer2(c, b1, W2, b2, m, hi, ebase, out);
}

// ---------------------------------------------------------------------------
// Fallback (f32 gather path): used only if ws_size can't hold the f16 copy.
// ---------------------------------------------------------------------------
__global__ __launch_bounds__(256) void VCSA_edge_mlp_f(
    const float* __restrict__ x,
    const int*   __restrict__ ei,
    const _Float16* __restrict__ wsB,
    const float* __restrict__ b1,
    const float* __restrict__ W2,
    const float* __restrict__ b2,
    float*       __restrict__ out)
{
    __shared__ _Float16 ldsB[W1_FRAG_ELEMS];
    const int lane = threadIdx.x & 31;
    const int wave = threadIdx.x >> 5;
    VCSA_tdm_fill_lds(wsB, wave, ldsB);

    const int tile  = blockIdx.x * 8 + wave;
    const int ebase = tile * 16;
    const int m     = lane & 15;
    const int hi    = (lane >= 16) ? 8 : 0;

    const int e    = ebase + m;
    const int nrow = ei[e];
    const int ncol = ei[N_EDGES_C + e];
    const float* __restrict__ xr = x + (size_t)nrow * NODE_DIM_C;
    const float* __restrict__ xc = x + (size_t)ncol * NODE_DIM_C;

    v8f c[4];
#pragma unroll
    for (int nt = 0; nt < 4; ++nt)
#pragma unroll
        for (int r = 0; r < 8; ++r) c[nt][r] = 0.0f;

#pragma unroll
    for (int kc = 0; kc < 8; ++kc) {
        const int k1 = kc * 32 + hi;
        const int k2 = k1 + 16;
        const float* p1 = (k1 < NODE_DIM_C) ? (xr + k1) : (xc + (k1 - NODE_DIM_C));
        const float* p2 = (k2 < NODE_DIM_C) ? (xr + k2) : (xc + (k2 - NODE_DIM_C));
        const float4 f0 = *(const float4*)(p1);
        const float4 f1 = *(const float4*)(p1 + 4);
        const float4 f2 = *(const float4*)(p2);
        const float4 f3 = *(const float4*)(p2 + 4);

        v16h a;
        a[0]  = (_Float16)f0.x; a[1]  = (_Float16)f0.y;
        a[2]  = (_Float16)f0.z; a[3]  = (_Float16)f0.w;
        a[4]  = (_Float16)f1.x; a[5]  = (_Float16)f1.y;
        a[6]  = (_Float16)f1.z; a[7]  = (_Float16)f1.w;
        a[8]  = (_Float16)f2.x; a[9]  = (_Float16)f2.y;
        a[10] = (_Float16)f2.z; a[11] = (_Float16)f2.w;
        a[12] = (_Float16)f3.x; a[13] = (_Float16)f3.y;
        a[14] = (_Float16)f3.z; a[15] = (_Float16)f3.w;

        v16h bfr[4];
#pragma unroll
        for (int nt = 0; nt < 4; ++nt) {
            const int fo = (nt * 8 + kc) * 512;
            const v8h blo = *(const v8h*)(ldsB + fo + lane * 8);
            const v8h bhi = *(const v8h*)(ldsB + fo + 256 + lane * 8);
            bfr[nt] = __builtin_shufflevector(blo, bhi,
                0, 1, 2, 3, 4, 5, 6, 7, 8, 9, 10, 11, 12, 13, 14, 15);
        }
#pragma unroll
        for (int nt = 0; nt < 4; ++nt) {
            c[nt] = __builtin_amdgcn_wmma_f32_16x16x32_f16(
                false, a, false, bfr[nt], (short)0, c[nt], false, false);
        }
    }
    VCSA_layer2(c, b1, W2, b2, m, hi, ebase, out);
}

// ---------------------------------------------------------------------------
extern "C" void kernel_launch(void* const* d_in, const int* in_sizes, int n_in,
                              void* d_out, int out_size, void* d_ws, size_t ws_size,
                              hipStream_t stream) {
    const float* x  = (const float*)d_in[0];
    const int*   ei = (const int*)  d_in[1];
    const float* W1 = (const float*)d_in[2];
    const float* b1 = (const float*)d_in[3];
    const float* W2 = (const float*)d_in[4];
    const float* b2 = (const float*)d_in[5];
    float* out = (float*)d_out;

    _Float16* wsB = (_Float16*)d_ws;                       // 32KB frag buffer
    _Float16* xh  = (_Float16*)((char*)d_ws + W1_FRAG_BYTES);

    VCSA_pack_w1<<<(W1_FRAG_ELEMS + 255) / 256, 256, 0, stream>>>(W1, wsB);

    const int tiles  = N_EDGES_C / 16;   // 50000, exact
    const int blocks = tiles / 8;        // 6250, 8 waves per block

    if (ws_size >= W1_FRAG_BYTES + XH_BYTES) {
        const int n8 = in_sizes[0] / 8;  // x elements / 8
        VCSA_pack_x<<<(n8 + 255) / 256, 256, 0, stream>>>(x, xh, n8);
        VCSA_edge_mlp_h<<<blocks, 256, 0, stream>>>(xh, ei, wsB, b1, W2, b2, out);
    } else {
        VCSA_edge_mlp_f<<<blocks, 256, 0, stream>>>(x, ei, wsB, b1, W2, b2, out);
    }
}
